// Type_Model_59133109732170
// MI455X (gfx1250) — matrix operand order repaired and checked by reading, not compiled, not run on hardware
//
#include <hip/hip_runtime.h>
#include <hip/hip_bf16.h>

// ---------------------------------------------------------------------------
// Problem constants (match reference)
// ---------------------------------------------------------------------------
#define SEQ_LEN   4096
#define INPUT_SZ  1024
#define HIDDEN    1024
#define GATES     (4 * HIDDEN)      // 4096
#define NUM_CLS   3

typedef __bf16 bf16;
typedef bf16  bf16x8  __attribute__((ext_vector_type(8)));
typedef bf16  bf16x16 __attribute__((ext_vector_type(16)));
typedef float floatx8 __attribute__((ext_vector_type(8)));
typedef unsigned int uint32x4 __attribute__((ext_vector_type(4)));
typedef int          int32x4  __attribute__((ext_vector_type(4)));
typedef int          int32x8  __attribute__((ext_vector_type(8)));

__device__ __forceinline__ float sigmoidf_dev(float x) {
    return 1.0f / (1.0f + __expf(-x));
}

// ---------------------------------------------------------------------------
// Zero-init (ws is poisoned 0xAA by harness; we need clean state + counters)
// ---------------------------------------------------------------------------
__global__ void zero_u32_kernel(unsigned* __restrict__ p, int n) {
    int i = blockIdx.x * blockDim.x + threadIdx.x;
    int stride = gridDim.x * blockDim.x;
    for (; i < n; i += stride) p[i] = 0u;
}

// ---------------------------------------------------------------------------
// fp32 -> bf16 conversion (WMMA feed precision)
// ---------------------------------------------------------------------------
__global__ void cvt_f32_bf16_kernel(const float* __restrict__ s,
                                    bf16* __restrict__ d, int n) {
    int i = blockIdx.x * blockDim.x + threadIdx.x;
    int stride = gridDim.x * blockDim.x;
    for (; i < n; i += stride) d[i] = (bf16)s[i];
}

// ---------------------------------------------------------------------------
// Input projection GEMM via WMMA:
//   C[m][n] = sum_k A[am][k] * B[n][k] + biasI[n] + biasH[n]
//   A : M x K row-major (X, bf16), optionally time-reversed (backward dir)
//   B : N x K row-major (Wih, bf16)  -> acts as WMMA B (KxN) since K contiguous
// Each wave computes a 16x64 output strip (4 accumulators) so the A fragment
// is reused across 4 WMMAs; 8 waves per block share the same A rows (L1 hit).
// ---------------------------------------------------------------------------
__global__ __launch_bounds__(256)
void gemm_xproj_wmma(const bf16* __restrict__ A,
                     const bf16* __restrict__ B,
                     const float* __restrict__ biasI,
                     const float* __restrict__ biasH,
                     float* __restrict__ C,
                     int M, int N, int K, int revA)
{
    const int lane  = threadIdx.x & 31;
    const int wave  = threadIdx.x >> 5;
    const int tileM = blockIdx.y;                    // 16-tall M tile per block
    const int nt0   = (blockIdx.x * 8 + wave) * 4;   // first of 4 N tiles
    const int mloc  = lane & 15;
    const int khalf = lane >> 4;                     // 0 or 1
    const int kbase = khalf * 16;                    // unused; A/B loads start at k0 + khalf*8

    int mrow = tileM * 16 + mloc;
    if (revA) mrow = M - 1 - mrow;

    const bf16* arow  = A + (size_t)mrow * K;
    const bf16* brow0 = B + (size_t)(nt0 * 16 + mloc) * K;

    floatx8 acc[4] = {{}, {}, {}, {}};
    for (int k0 = 0; k0 < K; k0 += 32) {
        // A 16x32 bf16 fragment (shared across the 4 N tiles)
        bf16x8 a0 = *reinterpret_cast<const bf16x8*>(arow + k0 + khalf * 8);
        bf16x8 a1 = *reinterpret_cast<const bf16x8*>(arow + k0 + 16 + khalf * 8);
        bf16x16 af;
#pragma unroll
        for (int e = 0; e < 8; ++e) { af[e] = a0[e]; af[8 + e] = a1[e]; }

#pragma unroll
        for (int t = 0; t < 4; ++t) {
            const bf16* bp = brow0 + (size_t)t * 16 * K + k0 + khalf * 8;
            bf16x8 b0 = *reinterpret_cast<const bf16x8*>(bp);
            bf16x8 b1 = *reinterpret_cast<const bf16x8*>(bp + 16);
            bf16x16 bm;
#pragma unroll
            for (int e = 0; e < 8; ++e) { bm[e] = b0[e]; bm[8 + e] = b1[e]; }
            // 8 args: (neg_a, A, neg_b, B, c_mod, C, reuse_a, reuse_b)
            acc[t] = __builtin_amdgcn_wmma_f32_16x16x32_bf16(
                         false, af, false, bm, (short)0, acc[t], false, false);
        }
    }

    // C/D layout: element v -> row tileM*16 + khalf*8 + v, col = ntile*16 + (lane&15)
#pragma unroll
    for (int t = 0; t < 4; ++t) {
        const int col  = (nt0 + t) * 16 + (lane & 15);
        const float bs = biasI[col] + biasH[col];
#pragma unroll
        for (int v = 0; v < 8; ++v) {
            int r = tileM * 16 + khalf * 8 + v;
            C[(size_t)r * N + col] = acc[t][v] + bs;
        }
    }
}

// ---------------------------------------------------------------------------
// Persistent LSTM recurrence. grid = (64 WGs, 2 directions), block = 256.
// WG w owns h indices [w*16, w*16+16) and the 64 matching gate rows
// (i/f/g/o at offsets 0,H,2H,3H). Whh slice lives in LDS (128 KB, needs
// CDNA5's 320KB/WGP), DMA'd in with the Tensor Data Mover.
// h double-buffered in global; one release/acquire grid barrier per timestep.
// ---------------------------------------------------------------------------
__global__ __launch_bounds__(256)
void lstm_recurrence(const float* __restrict__ xg_f,   // T x 4H
                     const float* __restrict__ xg_b,   // T x 4H (time-reversed proj)
                     const bf16*  __restrict__ whh_f,  // 4H x H bf16
                     const bf16*  __restrict__ whh_b,  // 4H x H bf16
                     float* __restrict__ hstate,       // [dir][2][H]
                     unsigned* __restrict__ counters)  // [dir]
{
    extern __shared__ unsigned char smem_raw[];

    const int dir = blockIdx.y;
    const int w   = blockIdx.x;          // 0..63
    const int tid = threadIdx.x;

    const float* xg  = dir ? xg_b : xg_f;
    const bf16*  Whh = dir ? whh_b : whh_f;
    float*    hbuf = hstate + (size_t)dir * 2 * HIDDEN;
    unsigned* cnt  = counters + dir;

    bf16*  s_w    = (bf16*)smem_raw;                  // 64 rows x 1024 bf16
    bf16*  s_h    = s_w + 64 * HIDDEN;                // 1024 bf16
    float* s_part = (float*)(s_h + HIDDEN);           // 256 f32
    float* s_gate = s_part + 256;                     // 64 f32
    float* s_c    = s_gate + 64;                      // 16 f32

#if __has_builtin(__builtin_amdgcn_tensor_load_to_lds)
    // TDM: DMA 4 strided 2D tiles (16 rows x 1024 bf16, row stride H) into LDS.
    // Issued by wave 0 only (TDM ignores EXEC; one issue per WG is enough).
    if (tid < 32) {
        const unsigned lds_base = (unsigned)(uintptr_t)s_w;  // flat[31:0] = LDS byte addr
#pragma unroll
        for (int g = 0; g < 4; ++g) {
            const unsigned long long gaddr =
                (unsigned long long)(uintptr_t)(Whh + (size_t)(g * HIDDEN + w * 16) * HIDDEN);
            uint32x4 d0;
            d0[0] = 1u;                                   // count=1, user-mode D#
            d0[1] = lds_base + (unsigned)(g * 16 * HIDDEN * 2);
            d0[2] = (unsigned)(gaddr & 0xffffffffu);      // global_addr[31:0]
            d0[3] = (unsigned)((gaddr >> 32) & 0x01ffffffu) | (2u << 30); // type=2
            int32x8 d1;
            d1[0] = (int)(1u << 16);          // wg_mask=0, data_size=1 (2 bytes)
            d1[1] = (int)(1024u << 16);       // tensor_dim0 = 1024 elements
            d1[2] = (int)(16u << 16);         // tensor_dim1 = 16 rows
            d1[3] = (int)(1024u << 16);       // tile_dim0 = 1024
            d1[4] = 16;                       // tile_dim1 = 16, tile_dim2 = 0
            d1[5] = 1024;                     // tensor_dim0_stride = 1024
            d1[6] = 0;
            d1[7] = 0;
            int32x4 d2 = {0, 0, 0, 0};
            int32x4 d3 = {0, 0, 0, 0};
            int32x8 d4 = {0, 0, 0, 0, 0, 0, 0, 0};
            // 6-arg form (clang-23 / therock-10.0 headers)
            __builtin_amdgcn_tensor_load_to_lds(d0, d1, d2, d3, d4, 0);
        }
        __builtin_amdgcn_s_wait_tensorcnt(0);
    }
#else
    // Fallback: cooperative VGPR copy of the Whh slice into LDS.
    for (int idx = tid; idx < 64 * (HIDDEN / 8); idx += 256) {
        int lr = idx / (HIDDEN / 8);
        int kc = (idx % (HIDDEN / 8)) * 8;
        int g = lr >> 4, j = lr & 15;
        const bf16x8* src = reinterpret_cast<const bf16x8*>(
            Whh + (size_t)(g * HIDDEN + w * 16 + j) * HIDDEN + kc);
        *reinterpret_cast<bf16x8*>(s_w + lr * HIDDEN + kc) = *src;
    }
#endif
    if (tid < 16) s_c[tid] = 0.0f;
    __syncthreads();

    const int lr   = tid >> 2;                               // 0..63 gate row
    const int q    = tid & 3;                                // quarter of K
    const int grow = (lr >> 4) * HIDDEN + w * 16 + (lr & 15);// global gate row

    for (int t = 0; t < SEQ_LEN; ++t) {
        // read h (double buffer t&1) -> LDS bf16
        const float* hin = hbuf + (t & 1) * HIDDEN;
        for (int k = tid; k < HIDDEN; k += 256) s_h[k] = (bf16)hin[k];
        __syncthreads();

        // 64 dot products of length 1024, fp32 accumulate from bf16 LDS
        const bf16* wr = s_w + lr * HIDDEN + q * 256;
        const bf16* hr = s_h + q * 256;
        float acc = 0.0f;
#pragma unroll 8
        for (int kk = 0; kk < 256; ++kk)
            acc += (float)wr[kk] * (float)hr[kk];
        s_part[tid] = acc;
        __syncthreads();

        if (q == 0) {
            float gsum = s_part[tid] + s_part[tid + 1] +
                         s_part[tid + 2] + s_part[tid + 3];
            s_gate[lr] = gsum + xg[(size_t)t * GATES + grow];
        }
        __syncthreads();

        if (tid < 16) {
            float iv = sigmoidf_dev(s_gate[tid]);
            float fv = sigmoidf_dev(s_gate[16 + tid]);
            float gv = tanhf(s_gate[32 + tid]);
            float ov = sigmoidf_dev(s_gate[48 + tid]);
            float c  = fv * s_c[tid] + iv * gv;
            s_c[tid] = c;
            hbuf[((t + 1) & 1) * HIDDEN + w * 16 + tid] = ov * tanhf(c);
        }
        __syncthreads();

        // grid barrier (release h writes, acquire before next read)
        if (tid == 0) {
            __hip_atomic_fetch_add(cnt, 1u, __ATOMIC_RELEASE,
                                   __HIP_MEMORY_SCOPE_AGENT);
            const unsigned target = 64u * (unsigned)(t + 1);
            while (__hip_atomic_load(cnt, __ATOMIC_ACQUIRE,
                                     __HIP_MEMORY_SCOPE_AGENT) < target)
                __builtin_amdgcn_s_sleep(1);
        }
        __syncthreads();
    }
}

// ---------------------------------------------------------------------------
// Output head: out[c] = dot(W_ho[c], [h_f, h_b]) + b_ho[c]
// Final h lives in buffer index 0 ((4095+1)&1 == 0).
// ---------------------------------------------------------------------------
__global__ __launch_bounds__(256)
void head_kernel(const float* __restrict__ hstate,   // [dir][2][H]
                 const float* __restrict__ W,        // 3 x 2H
                 const float* __restrict__ b,        // 3
                 float* __restrict__ out)            // 3
{
    __shared__ float red[256];
    const int c = blockIdx.x;
    const int tid = threadIdx.x;
    float s = 0.0f;
    for (int k = tid; k < 2 * HIDDEN; k += 256) {
        float hv = (k < HIDDEN) ? hstate[k]
                                : hstate[2 * HIDDEN + (k - HIDDEN)];
        s += W[(size_t)c * 2 * HIDDEN + k] * hv;
    }
    red[tid] = s;
    __syncthreads();
    for (int off = 128; off > 0; off >>= 1) {
        if (tid < off) red[tid] += red[tid + off];
        __syncthreads();
    }
    if (tid == 0) out[c] = red[0] + b[c];
}

// ---------------------------------------------------------------------------
// Host launcher
// ---------------------------------------------------------------------------
extern "C" void kernel_launch(void* const* d_in, const int* in_sizes, int n_in,
                              void* d_out, int out_size, void* d_ws, size_t ws_size,
                              hipStream_t stream) {
    (void)in_sizes; (void)n_in; (void)out_size; (void)ws_size;

    const float* inp   = (const float*)d_in[0];
    const float* Wih_f = (const float*)d_in[1];
    const float* Whh_f = (const float*)d_in[2];
    const float* bih_f = (const float*)d_in[3];
    const float* bhh_f = (const float*)d_in[4];
    const float* Wih_b = (const float*)d_in[5];
    const float* Whh_b = (const float*)d_in[6];
    const float* bih_b = (const float*)d_in[7];
    const float* bhh_b = (const float*)d_in[8];
    const float* W_ho  = (const float*)d_in[9];
    const float* b_ho  = (const float*)d_in[10];
    float* out = (float*)d_out;

    // workspace layout (all offsets 256B-aligned)
    unsigned char* ws = (unsigned char*)d_ws;
    const size_t sz_xbf  = (size_t)SEQ_LEN * INPUT_SZ * sizeof(bf16);  // 8 MB
    const size_t sz_wih  = (size_t)GATES * INPUT_SZ * sizeof(bf16);    // 8 MB
    const size_t sz_whh  = (size_t)GATES * HIDDEN * sizeof(bf16);      // 8 MB
    const size_t sz_xg   = (size_t)SEQ_LEN * GATES * sizeof(float);    // 64 MB

    bf16*  xbf    = (bf16*)(ws);
    bf16*  wihf_b = (bf16*)(ws + sz_xbf);
    bf16*  wihb_b = (bf16*)(ws + sz_xbf + sz_wih);
    bf16*  whhf_b = (bf16*)(ws + sz_xbf + 2 * sz_wih);
    bf16*  whhb_b = (bf16*)(ws + sz_xbf + 2 * sz_wih + sz_whh);
    float* xg_f   = (float*)(ws + sz_xbf + 2 * sz_wih + 2 * sz_whh);
    float* xg_b   = (float*)(ws + sz_xbf + 2 * sz_wih + 2 * sz_whh + sz_xg);
    float* hstate = (float*)(ws + sz_xbf + 2 * sz_wih + 2 * sz_whh + 2 * sz_xg);
    unsigned* counters = (unsigned*)(hstate + 2 * 2 * HIDDEN);

    // 1) zero h double-buffers (2 dirs x 2 bufs x 1024) + 2 barrier counters
    zero_u32_kernel<<<8, 256, 0, stream>>>((unsigned*)hstate,
                                           2 * 2 * HIDDEN + 2);

    // 2) fp32 -> bf16 conversions
    const int nX = SEQ_LEN * INPUT_SZ;
    const int nW = GATES * INPUT_SZ;
    cvt_f32_bf16_kernel<<<1024, 256, 0, stream>>>(inp,   xbf,    nX);
    cvt_f32_bf16_kernel<<<1024, 256, 0, stream>>>(Wih_f, wihf_b, nW);
    cvt_f32_bf16_kernel<<<1024, 256, 0, stream>>>(Wih_b, wihb_b, nW);
    cvt_f32_bf16_kernel<<<1024, 256, 0, stream>>>(Whh_f, whhf_b, nW);
    cvt_f32_bf16_kernel<<<1024, 256, 0, stream>>>(Whh_b, whhb_b, nW);

    // 3) input projections (WMMA GEMM): xg = X . Wih^T + (bih + bhh)
    //    each wave: 16x64 strip; block: 16 x 512; grid: (8, 256)
    dim3 ggrid(GATES / (16 * 4 * 8), SEQ_LEN / 16);
    gemm_xproj_wmma<<<ggrid, 256, 0, stream>>>(
        xbf, wihf_b, bih_f, bhh_f, xg_f, SEQ_LEN, GATES, INPUT_SZ, 0);
    gemm_xproj_wmma<<<ggrid, 256, 0, stream>>>(
        xbf, wihb_b, bih_b, bhh_b, xg_b, SEQ_LEN, GATES, INPUT_SZ, 1);

    // 4) persistent recurrence, both directions concurrently
    const size_t smem = 64 * HIDDEN * sizeof(bf16)   // Whh slice  (128 KB)
                      + HIDDEN * sizeof(bf16)        // h bf16
                      + 256 * sizeof(float)          // partials
                      + 64 * sizeof(float)           // gates
                      + 16 * sizeof(float);          // c state
    lstm_recurrence<<<dim3(64, 2), 256, smem, stream>>>(
        xg_f, xg_b, whhf_b, whhb_b, hstate, counters);

    // 5) head
    head_kernel<<<NUM_CLS, 256, 0, stream>>>(hstate, W_ho, b_ho, out);
}
